// HivemindMamba_52664888984004
// MI455X (gfx1250) — matrix operand order repaired and checked by reading, not compiled, not run on hardware
//
#include <hip/hip_runtime.h>
#include <cstdint>
#include <cstddef>

#define DM 512
#define DS 16
#define LT 2048
#define LI 1024
#define NB 2

typedef _Float16 v16h __attribute__((ext_vector_type(16)));
typedef _Float16 v8h  __attribute__((ext_vector_type(8)));
typedef float    v8f  __attribute__((ext_vector_type(8)));

union V16 { v16h v; v8h h[2]; };

// ---------------------------------------------------------------- conversions
__global__ void f32_to_f16_kernel(const float* __restrict__ in,
                                  _Float16* __restrict__ out, long n) {
  long i = (long)blockIdx.x * blockDim.x + threadIdx.x;
  long stride = (long)gridDim.x * blockDim.x;
  for (; i < n; i += stride) out[i] = (_Float16)in[i];
}

// f32 [R,C] row-major -> f16 [C,R] row-major (tiled via LDS)
__global__ void transpose_f32_to_f16_kernel(const float* __restrict__ in,
                                            _Float16* __restrict__ out,
                                            int R, int C) {
  __shared__ float tile[32][33];
  const int c0 = blockIdx.x * 32, r0 = blockIdx.y * 32;
  const int tx = threadIdx.x, ty = threadIdx.y;   // block (32, 8)
  for (int i = ty; i < 32; i += 8) {
    const int r = r0 + i, c = c0 + tx;
    if (r < R && c < C) tile[i][tx] = in[(long)r * C + c];
  }
  __syncthreads();
  for (int i = ty; i < 32; i += 8) {
    const int c = c0 + i, r = r0 + tx;
    if (c < C && r < R) out[(long)c * R + r] = (_Float16)tile[tx][i];
  }
}

// ---------------------------------------------------------------- WMMA GEMM
// C[M,N] = A[M,K](f16,row-major) * Bt^T (+bias)(+accum), Bt stored [N,K] f16.
// All fragment loads are contiguous 16B; no cross-register packing.
// Outputs (each optional): Cf (f32 [M,N]), Ch (f16 [M,N]), ChT (f16 [N,M],
//   written as one v8h store per lane per tile thanks to the C-layout).
// Each wave computes (16*MT)x(16*NT); block=(32,4); grid=(N/(16NT), M/(64MT), batch).
template <int MT, int NT>
__global__ void gemm_wmma_f16_kernel(
    const _Float16* __restrict__ A, int lda, long sA,
    const _Float16* __restrict__ Bt, int ldb, long sB,
    float* __restrict__ Cf, int ldc, long sC,
    _Float16* __restrict__ Ch, int ldch, long sCh,
    _Float16* __restrict__ ChT, int ldchT, long sChT,
    const float* __restrict__ bias,
    int M, int N, int K, int accum)
{
  const int lane = threadIdx.x;        // 0..31
  const int l16  = lane & 15;
  const int half = lane >> 4;
  const int m0 = (blockIdx.y * 4 + threadIdx.y) * (16 * MT);
  const int n0 = blockIdx.x * (16 * NT);
  if (m0 >= M || n0 >= N) return;      // wave-uniform
  const int z = blockIdx.z;

  // ISA 16-bit A 16x32 layout: lanes<16: elems0-7=K[k0..+7], 8-15=K[k0+16..+23];
  // lanes>=16: +8.  B 32x16 layout: same K order as A, N=n0+l16.
  const _Float16* Ab = A  + (long)z * sA + (long)(m0 + l16) * lda + 8 * half;
  const _Float16* Bb = Bt + (long)z * sB + (long)(n0 + l16) * ldb + 8 * half;

  v8f c[MT][NT];
#pragma unroll
  for (int mi = 0; mi < MT; ++mi)
#pragma unroll
    for (int ni = 0; ni < NT; ++ni) {
      if (accum) {
        const float* Cp = Cf + (long)z * sC;
#pragma unroll
        for (int r = 0; r < 8; ++r)
          c[mi][ni][r] =
              Cp[(long)(m0 + mi * 16 + r + 8 * half) * ldc + n0 + ni * 16 + l16];
      } else {
#pragma unroll
        for (int r = 0; r < 8; ++r) c[mi][ni][r] = 0.0f;
      }
    }

#pragma unroll 2
  for (int k0 = 0; k0 < K; k0 += 32) {
    V16 a[MT], b[NT];
#pragma unroll
    for (int mi = 0; mi < MT; ++mi) {
      a[mi].h[0] = *(const v8h*)(Ab + (long)mi * 16 * lda + k0);
      a[mi].h[1] = *(const v8h*)(Ab + (long)mi * 16 * lda + k0 + 16);
    }
#pragma unroll
    for (int ni = 0; ni < NT; ++ni) {
      const v8h* q = (const v8h*)(Bb + (long)ni * 16 * ldb + k0);
      b[ni].h[0] = q[0];
      b[ni].h[1] = q[2];
    }
#pragma unroll
    for (int mi = 0; mi < MT; ++mi)
#pragma unroll
      for (int ni = 0; ni < NT; ++ni)
        c[mi][ni] = __builtin_amdgcn_wmma_f32_16x16x32_f16(
            false, a[mi].v, false, b[ni].v, (short)0, c[mi][ni], false, false);
  }

  if (bias) {
#pragma unroll
    for (int ni = 0; ni < NT; ++ni) {
      const float bv = bias[n0 + ni * 16 + l16];
#pragma unroll
      for (int mi = 0; mi < MT; ++mi)
#pragma unroll
        for (int r = 0; r < 8; ++r) c[mi][ni][r] += bv;
    }
  }
  if (Cf) {
    float* Cp = Cf + (long)z * sC;
#pragma unroll
    for (int mi = 0; mi < MT; ++mi)
#pragma unroll
      for (int ni = 0; ni < NT; ++ni)
#pragma unroll
        for (int r = 0; r < 8; ++r)
          Cp[(long)(m0 + mi * 16 + r + 8 * half) * ldc + n0 + ni * 16 + l16] =
              c[mi][ni][r];
  }
  if (Ch) {
    _Float16* Hp = Ch + (long)z * sCh;
#pragma unroll
    for (int mi = 0; mi < MT; ++mi)
#pragma unroll
      for (int ni = 0; ni < NT; ++ni)
#pragma unroll
        for (int r = 0; r < 8; ++r)
          Hp[(long)(m0 + mi * 16 + r + 8 * half) * ldch + n0 + ni * 16 + l16] =
              (_Float16)c[mi][ni][r];
  }
  if (ChT) {
    // transposed f16 output [N, M]: per lane the 8 values are consecutive in M
    _Float16* Tp = ChT + (long)z * sChT;
#pragma unroll
    for (int mi = 0; mi < MT; ++mi)
#pragma unroll
      for (int ni = 0; ni < NT; ++ni) {
        v8h pk;
#pragma unroll
        for (int r = 0; r < 8; ++r) pk[r] = (_Float16)c[mi][ni][r];
        *(v8h*)(Tp + (long)(n0 + ni * 16 + l16) * ldchT + m0 + mi * 16 +
                8 * half) = pk;
      }
  }
}

// ---------------------------------------------------------------- softmax rows
__global__ void softmax_rows_kernel(const float* __restrict__ S,
                                    _Float16* __restrict__ O,
                                    int cols, float scale)
{
  const long row = blockIdx.x;
  const float* s = S + row * (long)cols;
  _Float16* o = O + row * (long)cols;
  __shared__ float red[8];
  const int tid = threadIdx.x, lane = tid & 31, wv = tid >> 5;

  float m = -3.0e38f;
  for (int i = tid; i < cols; i += 256) m = fmaxf(m, s[i]);
#pragma unroll
  for (int off = 16; off; off >>= 1) m = fmaxf(m, __shfl_xor(m, off, 32));
  if (lane == 0) red[wv] = m;
  __syncthreads();
  m = red[0];
#pragma unroll
  for (int i = 1; i < 8; ++i) m = fmaxf(m, red[i]);
  m *= scale;
  __syncthreads();

  float sum = 0.0f;
  for (int i = tid; i < cols; i += 256) sum += __expf(s[i] * scale - m);
#pragma unroll
  for (int off = 16; off; off >>= 1) sum += __shfl_xor(sum, off, 32);
  if (lane == 0) red[wv] = sum;
  __syncthreads();
  sum = 0.0f;
#pragma unroll
  for (int i = 0; i < 8; ++i) sum += red[i];
  const float inv = 1.0f / sum;
  for (int i = tid; i < cols; i += 256)
    o[i] = (_Float16)(__expf(s[i] * scale - m) * inv);
}

// ---------------------------------------------------------------- selective scan
// lane layout: s = lane&15, each wave owns 2 d-channels; block=256 -> 16 channels.
__global__ void selective_scan_kernel(
    const float* __restrict__ x, int ldx,
    const float* __restrict__ dt,
    const float* __restrict__ A_log,
    const float* __restrict__ Bm, const float* __restrict__ Cm,
    _Float16* __restrict__ yh,
    int L, int D)
{
  const int tid = threadIdx.x;
  const int lane = tid & 31, wave = tid >> 5;
  const int s = lane & 15;
  const int dflat = blockIdx.x * 16 + wave * 2 + (lane >> 4);
  const int b = dflat / D;
  const int d = dflat % D;

  const float Ads = -__expf(A_log[d * DS + s]);
  float h = 0.0f;
  const long base = (long)b * L;
  for (int l = 0; l < L; ++l) {
    const long idx = base + l;
    const float dtv = dt[idx * D + d];
    const float xv  = x[idx * (long)ldx + d];
    const float Bv  = Bm[idx * DS + s];
    const float Cv  = Cm[idx * DS + s];
    const float sp  = (dtv > 20.0f) ? dtv : log1pf(__expf(dtv));  // softplus
    h = __expf(sp * Ads) * h + sp * Bv * xv;
    float p = h * Cv;
#pragma unroll
    for (int off = 8; off; off >>= 1) p += __shfl_xor(p, off, 16);
    if (s == 0) yh[idx * D + d] = (_Float16)p;
  }
}

// ---------------------------------------------------------------- layernorm rows
__global__ void layernorm_rows_kernel(const float* __restrict__ X,
                                      const float* __restrict__ gamma,
                                      const float* __restrict__ beta,
                                      float* __restrict__ out)
{
  const long row = blockIdx.x;
  const float* xr = X + row * (long)DM;
  float* orow = out + row * (long)DM;
  __shared__ float rs[8], rs2[8];
  const int tid = threadIdx.x, lane = tid & 31, wv = tid >> 5;

  float s = 0.0f, s2 = 0.0f;
  for (int i = tid; i < DM; i += 256) { float v = xr[i]; s += v; s2 += v * v; }
#pragma unroll
  for (int off = 16; off; off >>= 1) {
    s  += __shfl_xor(s, off, 32);
    s2 += __shfl_xor(s2, off, 32);
  }
  if (lane == 0) { rs[wv] = s; rs2[wv] = s2; }
  __syncthreads();
  s = 0.0f; s2 = 0.0f;
#pragma unroll
  for (int i = 0; i < 8; ++i) { s += rs[i]; s2 += rs2[i]; }
  const float mu  = s * (1.0f / DM);
  const float var = s2 * (1.0f / DM) - mu * mu;
  const float inv = rsqrtf(var + 1e-5f);
  for (int i = tid; i < DM; i += 256)
    orow[i] = (xr[i] - mu) * inv * gamma[i] + beta[i];
}

// ---------------------------------------------------------------- launcher
extern "C" void kernel_launch(void* const* d_in, const int* in_sizes, int n_in,
                              void* d_out, int out_size, void* d_ws, size_t ws_size,
                              hipStream_t stream) {
  const float* text    = (const float*)d_in[0];
  const float* image   = (const float*)d_in[1];
  const float* W_t2i   = (const float*)d_in[2];
  const float* b_t2i   = (const float*)d_in[3];
  const float* W_i2t   = (const float*)d_in[4];
  const float* b_i2t   = (const float*)d_in[5];
  const float* W_dt_t  = (const float*)d_in[6];
  const float* b_dt_t  = (const float*)d_in[7];
  const float* W_dt_i  = (const float*)d_in[8];
  const float* b_dt_i  = (const float*)d_in[9];
  const float* W_B_t   = (const float*)d_in[10];
  const float* b_B_t   = (const float*)d_in[11];
  const float* W_B_i   = (const float*)d_in[12];
  const float* b_B_i   = (const float*)d_in[13];
  const float* W_C_t   = (const float*)d_in[14];
  const float* b_C_t   = (const float*)d_in[15];
  const float* W_C_i   = (const float*)d_in[16];
  const float* b_C_i   = (const float*)d_in[17];
  const float* A_log   = (const float*)d_in[18];
  const float* W_out   = (const float*)d_in[19];
  const float* b_out   = (const float*)d_in[20];
  const float* ln_g    = (const float*)d_in[21];
  const float* ln_b    = (const float*)d_in[22];
  float* out = (float*)d_out;

  const int Mt = NB * LT;   // 4096
  const int Mi = NB * LI;   // 2048

  // ---- workspace carve (256B aligned) ----
  char* ws = (char*)d_ws;
  size_t off = 0;
  auto take = [&](size_t bytes) -> void* {
    void* p = ws + off;
    off += (bytes + 255) & ~(size_t)255;
    return p;
  };
  _Float16* text_h   = (_Float16*)take((size_t)Mt * DM * 2);
  _Float16* img_h    = (_Float16*)take((size_t)Mi * DM * 2);
  _Float16* Wt2iT_h  = (_Float16*)take((size_t)DM * 2 * DM * 2); // [1024,512]
  _Float16* Wi2tT_h  = (_Float16*)take((size_t)DM * 2 * DM * 2); // [1024,512]
  _Float16* WdtT_t_h = (_Float16*)take((size_t)DM * DM * 2);     // [512,512]
  _Float16* WdtT_i_h = (_Float16*)take((size_t)DM * DM * 2);
  _Float16* WBT_t_h  = (_Float16*)take((size_t)DS * DM * 2);     // [16,512]
  _Float16* WBT_i_h  = (_Float16*)take((size_t)DS * DM * 2);
  _Float16* WCT_t_h  = (_Float16*)take((size_t)DS * DM * 2);
  _Float16* WCT_i_h  = (_Float16*)take((size_t)DS * DM * 2);
  _Float16* WoutT_h  = (_Float16*)take((size_t)DM * 2 * DM * 2); // [512,1024]
  float*    t2i_f    = (float*)take((size_t)Mt * 1024 * 4);
  _Float16* t2i_h    = (_Float16*)take((size_t)Mt * 1024 * 2);
  _Float16* t2iT_h   = (_Float16*)take((size_t)Mt * 1024 * 2);   // [1024, Mt]
  float*    i2t_f    = (float*)take((size_t)Mi * 1024 * 4);
  _Float16* i2t_h    = (_Float16*)take((size_t)Mi * 1024 * 2);
  _Float16* i2tT_h   = (_Float16*)take((size_t)Mi * 1024 * 2);   // [1024, Mi]
  float*    dt_t_f   = (float*)take((size_t)Mt * DM * 4);
  float*    dt_i_f   = (float*)take((size_t)Mi * DM * 4);
  float*    scores   = (float*)take((size_t)NB * LT * LI * 4);   // reused
  _Float16* attn_h   = (_Float16*)take((size_t)NB * LT * LI * 2);// reused
  _Float16* Bf_t_h   = (_Float16*)take((size_t)Mt * DM * 2);
  _Float16* Bf_i_h   = (_Float16*)take((size_t)Mi * DM * 2);
  float*    Bs_t     = (float*)take((size_t)Mt * DS * 4);
  float*    Cs_t     = (float*)take((size_t)Mt * DS * 4);
  float*    Bs_i     = (float*)take((size_t)Mi * DS * 4);
  float*    Cs_i     = (float*)take((size_t)Mi * DS * 4);
  _Float16* y_t_h    = (_Float16*)take((size_t)Mt * DM * 2);
  _Float16* y_i_h    = (_Float16*)take((size_t)Mi * DM * 2);
  float*    pre_t    = (float*)take((size_t)Mt * DM * 4);
  float*    pre_i    = (float*)take((size_t)Mi * DM * 4);

  auto conv = [&](const float* in, _Float16* o, long n) {
    int blocks = (int)((n + 255) / 256);
    if (blocks > 8192) blocks = 8192;
    f32_to_f16_kernel<<<blocks, 256, 0, stream>>>(in, o, n);
  };
  auto convT = [&](const float* in, _Float16* o, int R, int C) {
    dim3 g((C + 31) / 32, (R + 31) / 32), b(32, 8);
    transpose_f32_to_f16_kernel<<<g, b, 0, stream>>>(in, o, R, C);
  };

#define GEMM(MT_, NT_, Ap, lda, sA, Bp, ldb, sB, Cfp, ldc, sC, Chp, ldch, sCh,  \
             ChTp, ldchT, sChT, biasp, M, N, K, acc, batch)                      \
  gemm_wmma_f16_kernel<MT_, NT_>                                                 \
      <<<dim3((N) / (16 * (NT_)), (M) / (64 * (MT_)), (batch)), dim3(32, 4), 0,  \
         stream>>>(Ap, lda, sA, Bp, ldb, sB, Cfp, ldc, sC, Chp, ldch, sCh, ChTp, \
                   ldchT, sChT, biasp, M, N, K, acc)

  // ---- 0) conversions: activations row-major, all weights transposed ----
  conv(text,  text_h, (long)Mt * DM);
  conv(image, img_h,  (long)Mi * DM);
  convT(W_t2i, Wt2iT_h, DM, 2 * DM);
  convT(W_i2t, Wi2tT_h, DM, 2 * DM);
  convT(W_dt_t, WdtT_t_h, DM, DM);
  convT(W_dt_i, WdtT_i_h, DM, DM);
  convT(W_B_t, WBT_t_h, DM, DS);
  convT(W_B_i, WBT_i_h, DM, DS);
  convT(W_C_t, WCT_t_h, DM, DS);
  convT(W_C_i, WCT_i_h, DM, DS);
  convT(W_out, WoutT_h, 2 * DM, DM);

  // ---- 1) input projections (q|k fused; f32 + f16 + transposed-f16 out) ----
  GEMM(2, 2, text_h, DM, 0L, Wt2iT_h, DM, 0L, t2i_f, 1024, 0L,
       t2i_h, 1024, 0L, t2iT_h, Mt, 0L, b_t2i, Mt, 1024, DM, 0, 1);
  GEMM(2, 2, img_h, DM, 0L, Wi2tT_h, DM, 0L, i2t_f, 1024, 0L,
       i2t_h, 1024, 0L, i2tT_h, Mi, 0L, b_i2t, Mi, 1024, DM, 0, 1);

  // ---- 2) dt projections (A = q slice, cols 0..511 of 1024-wide f16) ----
  GEMM(2, 2, t2i_h, 1024, 0L, WdtT_t_h, DM, 0L, dt_t_f, DM, 0L,
       (_Float16*)nullptr, 0, 0L, (_Float16*)nullptr, 0, 0L,
       b_dt_t, Mt, DM, DM, 0, 1);
  GEMM(2, 2, i2t_h, 1024, 0L, WdtT_i_h, DM, 0L, dt_i_f, DM, 0L,
       (_Float16*)nullptr, 0, 0L, (_Float16*)nullptr, 0, 0L,
       b_dt_i, Mi, DM, DM, 0, 1);

  const float scale = 0.044194173824159216f; // 1/sqrt(512)

  // ---- 3) t2i attention: Q_t @ K_i^T -> softmax -> @ K_i (via K_i^T buf) ----
  GEMM(2, 2, t2i_h, 1024, (long)LT * 1024, i2t_h + DM, 1024, (long)LI * 1024,
       scores, LI, (long)LT * LI, (_Float16*)nullptr, 0, 0L,
       (_Float16*)nullptr, 0, 0L, (const float*)nullptr, LT, LI, DM, 0, NB);
  softmax_rows_kernel<<<NB * LT, 256, 0, stream>>>(scores, attn_h, LI, scale);
  GEMM(2, 2, attn_h, LI, (long)LT * LI, i2tT_h + (size_t)DM * Mi, Mi, (long)LI,
       (float*)nullptr, 0, 0L, Bf_t_h, DM, (long)LT * DM,
       (_Float16*)nullptr, 0, 0L, (const float*)nullptr, LT, DM, LI, 0, NB);

  // ---- 4) i2t attention (reuses scores/attn buffers) ----
  GEMM(2, 2, i2t_h, 1024, (long)LI * 1024, t2i_h + DM, 1024, (long)LT * 1024,
       scores, LT, (long)LI * LT, (_Float16*)nullptr, 0, 0L,
       (_Float16*)nullptr, 0, 0L, (const float*)nullptr, LI, LT, DM, 0, NB);
  softmax_rows_kernel<<<NB * LI, 256, 0, stream>>>(scores, attn_h, LT, scale);
  GEMM(2, 2, attn_h, LT, (long)LI * LT, t2iT_h + (size_t)DM * Mt, Mt, (long)LT,
       (float*)nullptr, 0, 0L, Bf_i_h, DM, (long)LI * DM,
       (_Float16*)nullptr, 0, 0L, (const float*)nullptr, LI, DM, LT, 0, NB);

  // ---- 5) B/C projections (N=16 single-tile-column GEMMs) ----
  GEMM(2, 1, Bf_t_h, DM, 0L, WBT_t_h, DM, 0L, Bs_t, DS, 0L,
       (_Float16*)nullptr, 0, 0L, (_Float16*)nullptr, 0, 0L,
       b_B_t, Mt, DS, DM, 0, 1);
  GEMM(2, 1, Bf_t_h, DM, 0L, WCT_t_h, DM, 0L, Cs_t, DS, 0L,
       (_Float16*)nullptr, 0, 0L, (_Float16*)nullptr, 0, 0L,
       b_C_t, Mt, DS, DM, 0, 1);
  GEMM(2, 1, Bf_i_h, DM, 0L, WBT_i_h, DM, 0L, Bs_i, DS, 0L,
       (_Float16*)nullptr, 0, 0L, (_Float16*)nullptr, 0, 0L,
       b_B_i, Mi, DS, DM, 0, 1);
  GEMM(2, 1, Bf_i_h, DM, 0L, WCT_i_h, DM, 0L, Cs_i, DS, 0L,
       (_Float16*)nullptr, 0, 0L, (_Float16*)nullptr, 0, 0L,
       b_C_i, Mi, DS, DM, 0, 1);

  // ---- 6) selective scans (x = q slice of f32 projection, ld 1024) ----
  selective_scan_kernel<<<NB * DM / 16, 256, 0, stream>>>(
      t2i_f, 1024, dt_t_f, A_log, Bs_t, Cs_t, y_t_h, LT, DM);
  selective_scan_kernel<<<NB * DM / 16, 256, 0, stream>>>(
      i2t_f, 1024, dt_i_f, A_log, Bs_i, Cs_i, y_i_h, LI, DM);

  // ---- 7) output projection: split-K handles the concat ----
  // WoutT is [512,1024]; K-halves are column slices 0..511 / 512..1023.
  GEMM(2, 2, text_h, DM, 0L, WoutT_h, 2 * DM, 0L, pre_t, DM, 0L,
       (_Float16*)nullptr, 0, 0L, (_Float16*)nullptr, 0, 0L,
       b_out, Mt, DM, DM, 0, 1);
  GEMM(2, 2, y_t_h, DM, 0L, WoutT_h + DM, 2 * DM, 0L, pre_t, DM, 0L,
       (_Float16*)nullptr, 0, 0L, (_Float16*)nullptr, 0, 0L,
       (const float*)nullptr, Mt, DM, DM, 1, 1);
  GEMM(2, 2, img_h, DM, 0L, WoutT_h, 2 * DM, 0L, pre_i, DM, 0L,
       (_Float16*)nullptr, 0, 0L, (_Float16*)nullptr, 0, 0L,
       b_out, Mi, DM, DM, 0, 1);
  GEMM(2, 2, y_i_h, DM, 0L, WoutT_h + DM, 2 * DM, 0L, pre_i, DM, 0L,
       (_Float16*)nullptr, 0, 0L, (_Float16*)nullptr, 0, 0L,
       (const float*)nullptr, Mi, DM, DM, 1, 1);

  // ---- 8) layernorm -> d_out (text block then image block, flat concat) ----
  layernorm_rows_kernel<<<Mt, 256, 0, stream>>>(pre_t, ln_g, ln_b, out);
  layernorm_rows_kernel<<<Mi, 256, 0, stream>>>(pre_i, ln_g, ln_b,
                                                out + (size_t)Mt * DM);
#undef GEMM
}
